// Model_88192858456176
// MI455X (gfx1250) — matrix operand order repaired and checked by reading, not compiled, not run on hardware
//
#include <hip/hip_runtime.h>

typedef __attribute__((ext_vector_type(16))) _Float16 v16h;
typedef __attribute__((ext_vector_type(8)))  float    v8f;

#define DD        128          // feature / hidden dim
#define TT        32           // time dim
#define KXX       160          // DD + TT (GRU input dim)
#define NG        384          // 3*DD gate rows
#define NBLK_GI   120          // 24 row-tiles * 5 K-tiles (K=160)
#define NBLK_GH   96           // 24 row-tiles * 4 K-tiles (K=128)
#define WPACK_ELEMS ((NBLK_GI + NBLK_GH) * 512)   // 110592 f16
#define ECAP      2048
#define NTHREADS  384          // 12 waves (wave32)

// ---------------------------------------------------------------------------
// Pack W_ih / W_hh into f16 A-matrix tile layout for v_wmma_f32_16x16x32_f16.
// Per 16x32 tile: 512 f16 stored as [lane][16 halfwords]; element q = 2v+h of
// lane l maps to M = l&15,  K = (l<16) ? (v<4 ? 2v+h : 16+2(v-4)+h)
//                                      : (v<4 ? 8+2v+h : 24+2(v-4)+h)
// ---------------------------------------------------------------------------
__global__ void pack_weights_k(const float* __restrict__ W_ih,
                               const float* __restrict__ W_hh,
                               _Float16* __restrict__ wpack) {
  int idx = blockIdx.x * blockDim.x + threadIdx.x;
  if (idx >= WPACK_ELEMS) return;
  int b = idx >> 9;        // tile-block index
  int r = idx & 511;
  int l = r >> 4;          // lane 0..31
  int q = r & 15;          // halfword within lane
  int v = q >> 1;
  int h = q & 1;
  int M = l & 15;
  int K;
  if (l < 16) K = (v < 4) ? (2 * v + h) : (16 + 2 * (v - 4) + h);
  else        K = (v < 4) ? (8 + 2 * v + h) : (24 + 2 * (v - 4) + h);
  float val;
  if (b < NBLK_GI) {
    int tt = b / 5, kt = b % 5;
    val = W_ih[(tt * 16 + M) * KXX + (kt * 32 + K)];
  } else {
    int b2 = b - NBLK_GI;
    int tt = b2 / 4, kt = b2 % 4;
    val = W_hh[(tt * 16 + M) * DD + (kt * 32 + K)];
  }
  wpack[idx] = (_Float16)val;
}

__global__ void init_map_k(int* __restrict__ map, int n) {
  int i = blockIdx.x * blockDim.x + threadIdx.x;
  if (i < n) map[i] = -1;
}

__device__ __forceinline__ v16h splat16(_Float16 x) {
  v16h v;
#pragma unroll
  for (int u = 0; u < 16; ++u) v[u] = x;
  return v;
}

// ---------------------------------------------------------------------------
// Sequential edge-scan GRU: one persistent workgroup (12 waves).
// Weights live in VGPRs; edges/biases/t2v params staged in LDS once.
// Row versioning: map[node] = latest slot in upd[] (or -1 => raw features).
// 4 barriers per step; next-edge map lookups software-pipelined into the tail.
// ---------------------------------------------------------------------------
__launch_bounds__(NTHREADS, 1)
__global__ void gru_seq_k(const float* __restrict__ raw,
                          const int* __restrict__ edges, int E,
                          const float* __restrict__ t2v_w0, const float* __restrict__ t2v_b0,
                          const float* __restrict__ t2v_w,  const float* __restrict__ t2v_b,
                          const float* __restrict__ b_ih,   const float* __restrict__ b_hh,
                          const _Float16* __restrict__ wpack,
                          int* __restrict__ map, float* __restrict__ upd) {
  __shared__ int   eds[ECAP * 3];
  __shared__ float xf[KXX];      // raw src (0..127) ++ time2vec (128..159)
  __shared__ float hf[DD];       // target hidden
  __shared__ float gif[NG];      // W_ih @ x
  __shared__ float ghf[NG];      // W_hh @ h
  __shared__ float bihf[NG];
  __shared__ float bhhf[NG];
  __shared__ float tvw[TT];
  __shared__ float tvb[TT];
  __shared__ float red[16];
  __shared__ int   einfo[4];

  const int tid  = threadIdx.x;
  const int wave = tid >> 5;
  const int lane = tid & 31;

  // ---- one-time staging ----
  const int ecached = (E < ECAP ? E : ECAP) * 3;
  for (int i = tid; i < ecached; i += NTHREADS) eds[i] = edges[i];
  for (int i = tid; i < NG; i += NTHREADS) { bihf[i] = b_ih[i]; bhhf[i] = b_hh[i]; }
  if (tid < TT - 1) { tvw[tid] = t2v_w[tid]; tvb[tid] = t2v_b[tid]; }
  if (tid == 0)     { tvw[TT - 1] = t2v_w0[0]; tvb[TT - 1] = t2v_b0[0]; }

  // weights -> registers: per wave 2 gi row-tiles (5 K-tiles) + 2 gh (4 K-tiles)
  v16h agi[2][5];
  v16h agh[2][4];
#pragma unroll
  for (int j = 0; j < 2; ++j) {
    const int tt = wave * 2 + j;
#pragma unroll
    for (int kt = 0; kt < 5; ++kt)
      agi[j][kt] = *(const v16h*)(wpack + (size_t)((tt * 5 + kt) * 512 + lane * 16));
#pragma unroll
    for (int kt = 0; kt < 4; ++kt)
      agh[j][kt] = *(const v16h*)(wpack + (size_t)((NBLK_GI + tt * 4 + kt) * 512 + lane * 16));
  }

  // prologue: map lookups for edge 0
  if (tid == 0 && E > 0) {
    einfo[0] = map[edges[0]];
    einfo[1] = map[edges[1]];
  }
  __syncthreads();

  for (int e = 0; e < E; ++e) {
    int s, t, tmi;
    if (e < ECAP) { s = eds[3 * e]; t = eds[3 * e + 1]; tmi = eds[3 * e + 2]; }
    else          { s = edges[3 * e]; t = edges[3 * e + 1]; tmi = edges[3 * e + 2]; }
    const float tm = (float)tmi;
    const int ms = einfo[0], mt = einfo[1];
    const float* __restrict__ srcp = (ms < 0) ? (raw + (size_t)s * DD) : (upd + (size_t)ms * DD);
    const float* __restrict__ hp   = (mt < 0) ? (raw + (size_t)t * DD) : (upd + (size_t)mt * DD);

    float sval = 0.0f;
    if (tid < DD) {                      // waves 0-3: load src, wave-reduce ||src||^2
      sval = srcp[tid];
      xf[tid] = sval;                    // raw; inv_s folded into B build
      float v = sval * sval;
#pragma unroll
      for (int m = 16; m > 0; m >>= 1) v += __shfl_xor(v, m, 32);
      if (lane == 0) red[wave] = v;
    } else if (tid < 2 * DD) {           // waves 4-7: load h, compute time2vec
      const int j = tid - DD;
      hf[j] = hp[j];
      if (j < TT)
        xf[DD + j] = (j < TT - 1) ? sinf(tm * tvw[j] + tvb[j])
                                  : (tm * tvw[TT - 1] + tvb[TT - 1]);
    }
    __syncthreads();                                            // B_a
    const float inv_s = 1.0f / fmaxf(sqrtf(red[0] + red[1] + red[2] + red[3]), 1e-12f);

    // ---- WMMA matvecs: B built once per K-tile, shared by both row-tiles ----
    const int tt0 = wave * 2, tt1 = wave * 2 + 1;
    {
      v8f c0 = {}, c1 = {};
#pragma unroll
      for (int kt = 0; kt < 5; ++kt) {
        v16h b;
#pragma unroll
        for (int u = 0; u < 16; ++u) {
          const int kk = (u < 8) ? (8 * (lane >> 4) + u) : (16 + 8 * (lane >> 4) + (u - 8));
          float xv = xf[(kt << 5) + kk];
          if (kt < 4) xv *= inv_s;         // normalize src lanes; t2v lanes raw
          b[u] = (_Float16)xv;
        }
        c0 = __builtin_amdgcn_wmma_f32_16x16x32_f16(false, agi[0][kt], false, b,
                                                    (short)0, c0, false, false);
        c1 = __builtin_amdgcn_wmma_f32_16x16x32_f16(false, agi[1][kt], false, b,
                                                    (short)0, c1, false, false);
      }
      if (lane == 0) {
#pragma unroll
        for (int u = 0; u < 8; ++u) { gif[tt0 * 16 + u] = c0[u]; gif[tt1 * 16 + u] = c1[u]; }
      } else if (lane == 16) {
#pragma unroll
        for (int u = 0; u < 8; ++u) { gif[tt0 * 16 + 8 + u] = c0[u]; gif[tt1 * 16 + 8 + u] = c1[u]; }
      }
    }
    {
      v8f d0 = {}, d1 = {};
#pragma unroll
      for (int kt = 0; kt < 4; ++kt) {
        v16h b;
#pragma unroll
        for (int u = 0; u < 16; ++u)
          b[u] = (_Float16)hf[(kt << 5) + ((u < 8) ? (8 * (lane >> 4) + u) : (16 + 8 * (lane >> 4) + (u - 8)))];
        d0 = __builtin_amdgcn_wmma_f32_16x16x32_f16(false, agh[0][kt], false, b,
                                                    (short)0, d0, false, false);
        d1 = __builtin_amdgcn_wmma_f32_16x16x32_f16(false, agh[1][kt], false, b,
                                                    (short)0, d1, false, false);
      }
      if (lane == 0) {
#pragma unroll
        for (int u = 0; u < 8; ++u) { ghf[tt0 * 16 + u] = d0[u]; ghf[tt1 * 16 + u] = d1[u]; }
      } else if (lane == 16) {
#pragma unroll
        for (int u = 0; u < 8; ++u) { ghf[tt0 * 16 + 8 + u] = d0[u]; ghf[tt1 * 16 + 8 + u] = d1[u]; }
      }
    }
    __syncthreads();                                            // B_b

    // ---- gates + new hidden + its norm ----
    float hn = 0.0f;
    if (tid < DD) {
      const float a_r = gif[tid]          + bihf[tid]          + ghf[tid]          + bhhf[tid];
      const float a_z = gif[DD + tid]     + bihf[DD + tid]     + ghf[DD + tid]     + bhhf[DD + tid];
      const float rr = 1.0f / (1.0f + expf(-a_r));
      const float zz = 1.0f / (1.0f + expf(-a_z));
      const float nn = tanhf(gif[2 * DD + tid] + bihf[2 * DD + tid] +
                             rr * (ghf[2 * DD + tid] + bhhf[2 * DD + tid]));
      hn = (1.0f - zz) * nn + zz * hf[tid];
      float v = hn * hn;
#pragma unroll
      for (int m = 16; m > 0; m >>= 1) v += __shfl_xor(v, m, 32);
      if (lane == 0) red[wave] = v;
    }
    __syncthreads();                                            // B_c
    const float inv_h = 1.0f / fmaxf(sqrtf(red[0] + red[1] + red[2] + red[3]), 1e-12f);
    if (tid < DD) upd[(size_t)e * DD + tid] = hn * inv_h;
    if (tid == 0) {
      map[t] = e;                        // version bump for target row
      if (e + 1 < E) {                   // pipeline next edge's map lookups
        int s1, t1;
        if (e + 1 < ECAP) { s1 = eds[3 * (e + 1)]; t1 = eds[3 * (e + 1) + 1]; }
        else              { s1 = edges[3 * (e + 1)]; t1 = edges[3 * (e + 1) + 1]; }
        einfo[0] = map[s1];              // same-thread store->load: sees map[t]=e
        einfo[1] = map[t1];
      }
    }
    __threadfence();                     // make upd/map visible block-wide
    __syncthreads();                                            // B_d
  }
}

// out = tanh(latest feats), with row indirection instead of a full feats copy.
__global__ void finalize_k(const float* __restrict__ raw,
                           const float* __restrict__ upd,
                           const int* __restrict__ map,
                           float* __restrict__ out, long long nd) {
  long long i = (long long)blockIdx.x * blockDim.x + threadIdx.x;
  if (i >= nd) return;
  const int row = (int)(i >> 7);
  const int m   = map[row];
  const float v = (m < 0) ? raw[i] : upd[(size_t)m * DD + (int)(i & 127)];
  out[i] = tanhf(v);
}

extern "C" void kernel_launch(void* const* d_in, const int* in_sizes, int n_in,
                              void* d_out, int out_size, void* d_ws, size_t ws_size,
                              hipStream_t stream) {
  const float* raw    = (const float*)d_in[0];
  const int*   edges  = (const int*)d_in[1];
  const float* t2v_w0 = (const float*)d_in[2];
  const float* t2v_b0 = (const float*)d_in[3];
  const float* t2v_w  = (const float*)d_in[4];
  const float* t2v_b  = (const float*)d_in[5];
  const float* W_ih   = (const float*)d_in[6];
  const float* W_hh   = (const float*)d_in[7];
  const float* b_ih   = (const float*)d_in[8];
  const float* b_hh   = (const float*)d_in[9];
  const int N = in_sizes[0] / DD;
  const int E = in_sizes[1] / 3;
  float* out = (float*)d_out;

  // workspace layout: [wpack f16 | map int[N] | upd float[E*DD]]
  char* ws = (char*)d_ws;
  _Float16* wpack = (_Float16*)ws;
  size_t off = (size_t)WPACK_ELEMS * sizeof(_Float16);
  int* map = (int*)(ws + off);
  off += (size_t)N * sizeof(int);
  off = (off + 255) & ~(size_t)255;
  float* upd = (float*)(ws + off);

  pack_weights_k<<<(WPACK_ELEMS + 255) / 256, 256, 0, stream>>>(W_ih, W_hh, wpack);
  init_map_k<<<(N + 255) / 256, 256, 0, stream>>>(map, N);
  gru_seq_k<<<1, NTHREADS, 0, stream>>>(raw, edges, E, t2v_w0, t2v_b0, t2v_w, t2v_b,
                                        b_ih, b_hh, wpack, map, upd);
  const long long nd = (long long)N * DD;
  finalize_k<<<(int)((nd + 255) / 256), 256, 0, stream>>>(raw, upd, map, out, nd);
}
